// AxialAttentionBlock_72885595013399
// MI455X (gfx1250) — matrix operand order repaired and checked by reading, not compiled, not run on hardware
//
#include <hip/hip_runtime.h>
#include <hip/hip_bf16.h>
#include <math.h>

// ---------------------------------------------------------------------------
// AxialAttentionBlock for MI455X (gfx1250, wave32, WMMA 16x16x32 bf16).
// bf16 inputs / fp32 accumulation; async global->LDS double-buffered GEMMs.
// ---------------------------------------------------------------------------

typedef __attribute__((ext_vector_type(16))) __bf16 v16bf;
typedef __attribute__((ext_vector_type(8)))  float  v8f;
typedef __attribute__((ext_vector_type(4)))  int    v4i_t;

#define TOKS  32768          // B*H*W tokens
#define CH    768
#define NHE   12
#define HDIM  64
#define HWSP  4096           // H*W
#define NB    8
#define SLABS 64             // partial-reduction slabs for instance norm
#define LDQ   72             // LDS row stride (bf16) for attention tiles
#define GSTR  72             // LDS row stride for 64-wide GEMM tiles (144B)
#define BKK   64             // K per staged tile (2 WMMA k-substeps)

#if defined(__gfx1250__) && \
    __has_builtin(__builtin_amdgcn_global_load_async_to_lds_b128) && \
    __has_builtin(__builtin_amdgcn_s_wait_asynccnt)
#define HAVE_ASYNC 1
#define ASYNC_WAIT(n) __builtin_amdgcn_s_wait_asynccnt(n)
#else
#define HAVE_ASYNC 0
#define ASYNC_WAIT(n)
#endif

// Copy 16 bytes global -> LDS (async DMA when available).
// Builtin prototype (probe-confirmed): (v4i32 AS1*, v4i32 AS3*, imm off, imm cpol)
static __device__ __forceinline__ void cp16(const __bf16* g, __bf16* l) {
#if HAVE_ASYNC
  __builtin_amdgcn_global_load_async_to_lds_b128(
      (__attribute__((address_space(1))) v4i_t*)(g),
      (__attribute__((address_space(3))) v4i_t*)(l), 0, 0);
#else
  *(float4*)l = *(const float4*)g;
#endif
}

static __device__ __forceinline__ __bf16 f2bf(float f) {
  unsigned u = __builtin_bit_cast(unsigned, f);
  unsigned r = u + 0x7fffu + ((u >> 16) & 1u);     // round-to-nearest-even
  unsigned short h = (unsigned short)(r >> 16);
  return __builtin_bit_cast(__bf16, h);
}

union FragU { v16bf v; unsigned u[8]; };

// 16-bit A-matrix 16x32 fragment (ISA 7.12.2): lanes 0-15 -> M=lane, K groups
// {0..7,16..23}; lanes 16-31 -> M=lane-16, K groups {8..15,24..31}.
static __device__ __forceinline__ v16bf load_a_frag(const __bf16* base, int ld, int lane) {
  const int m  = lane & 15;
  const int kb = (lane < 16) ? 0 : 8;
  const __bf16* row = base + m * ld;
  FragU f;
#pragma unroll
  for (int i = 0; i < 8; ++i) {
    const int k = (i < 4) ? (kb + 2 * i) : (kb + 16 + 2 * (i - 4));
    f.u[i] = *(const unsigned*)(row + k);          // bf16 pair (k, k+1)
  }
  return f.v;
}

// 16-bit B-matrix 32x16 fragment from [n][k] storage: lanes 0-15 hold K groups
// {0..7,16..23} of column N=lane, lanes 16-31 hold K groups {8..15,24..31}.
static __device__ __forceinline__ v16bf load_bt_frag(const __bf16* base, int ld, int lane) {
  const int n  = lane & 15;
  const int kb = (lane < 16) ? 0 : 8;
  const __bf16* row = base + n * ld;
  FragU f;
#pragma unroll
  for (int i = 0; i < 8; ++i) {
    const int k = (i < 4) ? (kb + 2 * i) : (kb + 16 + 2 * (i - 4));
    f.u[i] = *(const unsigned*)(row + k);
  }
  return f.v;
}

static __device__ __forceinline__ v8f wmma_bf16(v16bf a, v16bf b, v8f c) {
  return __builtin_amdgcn_wmma_f32_16x16x32_bf16(false, a, false, b, (short)0, c,
                                                 false, false);
}

// Reductions over the 16-lane half-groups (row of a C/D fragment).
static __device__ __forceinline__ float hmax16(float v) {
  v = fmaxf(v, __shfl_xor(v, 1, 32));
  v = fmaxf(v, __shfl_xor(v, 2, 32));
  v = fmaxf(v, __shfl_xor(v, 4, 32));
  v = fmaxf(v, __shfl_xor(v, 8, 32));
  return v;
}
static __device__ __forceinline__ float hsum16(float v) {
  v += __shfl_xor(v, 1, 32);
  v += __shfl_xor(v, 2, 32);
  v += __shfl_xor(v, 4, 32);
  v += __shfl_xor(v, 8, 32);
  return v;
}

static __device__ __forceinline__ float gelu_tanh(float x) {
  const float x3 = x * x * x;
  return 0.5f * x * (1.0f + tanhf(0.7978845608028654f * (x + 0.044715f * x3)));
}

// ---------------------------------------------------------------------------
// Elementwise / reduction kernels
// ---------------------------------------------------------------------------

// d[n*K + k] = bf16(s[k*N + n])  (weight transpose to n-major for GEMM B tiles)
__global__ __launch_bounds__(256) void transpose_to_bf16_kernel(
    const float* __restrict__ s, __bf16* __restrict__ d, int K, int N) {
  const long i = (long)blockIdx.x * 256 + threadIdx.x;
  if (i >= (long)K * N) return;
  const int k = (int)(i % K);
  const int n = (int)(i / K);
  d[i] = f2bf(s[(long)k * N + n]);
}

__global__ __launch_bounds__(256) void build_bias_kernel(
    const float* __restrict__ rel, float* __restrict__ bt) {
  const int i  = blockIdx.x * 256 + threadIdx.x;   // < 12*64*64
  const int e  = i >> 12;
  const int r  = (i >> 6) & 63;
  const int cc = i & 63;
  bt[i] = rel[(r - cc + 63) * NHE + e];
}

// Instance-norm std (ddof=1) over spatial, stage 1: per-slab partial sums.
// LAYOUT 0: (B,HW,C) source.  LAYOUT 1: (B,HE,HW,HD) source.
template<int LAYOUT>
__global__ __launch_bounds__(256) void inorm_partial_kernel(
    const float* __restrict__ src, float* __restrict__ p1,
    float* __restrict__ p2, float pre) {
  const int c  = blockIdx.x * 256 + threadIdx.x;
  const int b  = blockIdx.y;
  const int sl = blockIdx.z;
  float s1 = 0.f, s2 = 0.f;
  const int hw0 = sl * (HWSP / SLABS);
  for (int hw = hw0; hw < hw0 + HWSP / SLABS; ++hw) {
    long idx;
    if (LAYOUT == 0) {
      idx = ((long)b * HWSP + hw) * CH + c;
    } else {
      const int he = c >> 6, d = c & 63;
      idx = ((long)(b * NHE + he) * HWSP + hw) * HDIM + d;
    }
    const float v = src[idx] * pre;
    s1 += v; s2 += v * v;
  }
  const long o = ((long)sl * NB + b) * CH + c;
  p1[o] = s1; p2[o] = s2;
}

__global__ __launch_bounds__(256) void inorm_finish_kernel(
    const float* __restrict__ p1, const float* __restrict__ p2,
    const float* __restrict__ w, float* __restrict__ scale,
    float pre, float eps) {
  const int c = blockIdx.x * 256 + threadIdx.x;
  const int b = blockIdx.y;
  float s1 = 0.f, s2 = 0.f;
  for (int sl = 0; sl < SLABS; ++sl) {
    const long o = ((long)sl * NB + b) * CH + c;
    s1 += p1[o]; s2 += p2[o];
  }
  const float nn  = (float)HWSP;
  const float var = (s2 - s1 * s1 / nn) / (nn - 1.0f);
  scale[b * CH + c] = pre * w[c] / (sqrtf(fmaxf(var, 0.f)) + eps);
}

template<int LAYOUT>
__global__ __launch_bounds__(256) void inorm_apply_kernel(
    const float* __restrict__ src, const float* __restrict__ scale,
    __bf16* __restrict__ dst) {
  const long i   = (long)blockIdx.x * 256 + threadIdx.x;   // token-major dst index
  const int  c   = (int)(i % CH);
  const long tok = i / CH;
  const int  b   = (int)(tok >> 12);
  long sidx;
  if (LAYOUT == 0) {
    sidx = i;
  } else {
    const int hw = (int)(tok & 4095);
    const int he = c >> 6, d = c & 63;
    sidx = ((long)(b * NHE + he) * HWSP + hw) * HDIM + d;
  }
  dst[i] = f2bf(src[sidx] * scale[b * CH + c]);
}

__global__ __launch_bounds__(256) void final_kernel(
    const float* __restrict__ m, const float* __restrict__ scale3,
    const float* __restrict__ gmlp, const float* __restrict__ x1,
    float* __restrict__ out) {
  const long i = (long)blockIdx.x * 256 + threadIdx.x;
  const int  c = (int)(i % CH);
  const int  b = (int)(i / ((long)HWSP * CH));
  out[i] = m[i] * scale3[b * CH + c] * gmlp[c] + x1[i];
}

// ---------------------------------------------------------------------------
// bf16 GEMM: C(MxN) = A(MxK) @ Bt^T where Bt is the (NxK) transposed weight.
// 128x128 tile, BK=64 stages (2 WMMA k-substeps), 8 waves, double-buffered
// async global->LDS staging (8 B128 DMA ops in flight per thread).
// EPI 0: out_f32 = acc + bias
// EPI 1: out_bf16 = gelu(acc + bias)
// EPI 2: r = (acc + bias)*gamma + res; out_f32 = r; out_bf16 = r
// ---------------------------------------------------------------------------
template<int EPI>
__global__ __launch_bounds__(256) void gemm_bf16_kernel(
    const __bf16* __restrict__ A, const __bf16* __restrict__ Bt,
    const float* __restrict__ bias, const float* __restrict__ gamma,
    const float* __restrict__ res, float* __restrict__ outf,
    __bf16* __restrict__ outb, int M, int N, int K) {
  __shared__ __bf16 As[2][128 * GSTR];
  __shared__ __bf16 Bs[2][128 * GSTR];   // [n][k] tile, same geometry as A
  const int tid  = threadIdx.x;
  const int lane = tid & 31;
  const int wv   = tid >> 5;
  const int wm   = wv & 3;               // 4 waves along M (32 rows each)
  const int wn   = wv >> 2;              // 2 waves along N (64 cols each)
  const int m0   = blockIdx.y * 128;
  const int n0   = blockIdx.x * 128;

  auto issue_tile = [&](int k0, int buf) {
#pragma unroll
    for (int i = 0; i < 4; ++i) {        // 1024 x 16B chunks per matrix tile
      const int ch = tid + i * 256;
      const int r = ch >> 3, cc = ch & 7;
      cp16(A  + (long)(m0 + r) * K + k0 + cc * 8, &As[buf][r * GSTR + cc * 8]);
      cp16(Bt + (long)(n0 + r) * K + k0 + cc * 8, &Bs[buf][r * GSTR + cc * 8]);
    }
  };

  const v8f vz = {0.f, 0.f, 0.f, 0.f, 0.f, 0.f, 0.f, 0.f};
  v8f acc[2][4];
#pragma unroll
  for (int s = 0; s < 2; ++s)
#pragma unroll
    for (int t = 0; t < 4; ++t) acc[s][t] = vz;

  const int KT = K / BKK;
  issue_tile(0, 0);
#pragma clang loop unroll(disable)
  for (int kt = 0; kt < KT; ++kt) {
    const int cur = kt & 1;
    if (kt + 1 < KT) {
      issue_tile((kt + 1) * BKK, cur ^ 1);
      ASYNC_WAIT(8);                     // in-order: previous tile's 8 ops done
    } else {
      ASYNC_WAIT(0);
    }
    __syncthreads();
#pragma unroll
    for (int ks = 0; ks < 2; ++ks) {     // two 32-deep WMMA substeps
      const int ko = ks * 32;
      v16bf a0 = load_a_frag(&As[cur][(wm * 32) * GSTR + ko], GSTR, lane);
      v16bf a1 = load_a_frag(&As[cur][(wm * 32 + 16) * GSTR + ko], GSTR, lane);
#pragma unroll
      for (int t = 0; t < 4; ++t) {
        v16bf b = load_bt_frag(&Bs[cur][(wn * 64 + t * 16) * GSTR + ko], GSTR, lane);
        acc[0][t] = wmma_bf16(a0, b, acc[0][t]);
        acc[1][t] = wmma_bf16(a1, b, acc[1][t]);
      }
    }
    __syncthreads();
  }

#pragma unroll
  for (int s = 0; s < 2; ++s)
#pragma unroll
    for (int j = 0; j < 8; ++j) {
      const int m = m0 + wm * 32 + s * 16 + j + ((lane & 16) ? 8 : 0);
#pragma unroll
      for (int t = 0; t < 4; ++t) {
        const int  n   = n0 + wn * 64 + t * 16 + (lane & 15);
        const long idx = (long)m * N + n;
        float v = acc[s][t][j] + bias[n];
        if (EPI == 0) {
          outf[idx] = v;
        } else if (EPI == 1) {
          outb[idx] = f2bf(gelu_tanh(v));
        } else {
          float r = v * gamma[n] + res[idx];
          outf[idx] = r;
          outb[idx] = f2bf(r);
        }
      }
    }
}

// ---------------------------------------------------------------------------
// QKV GEMM: wave N-span (64) == one head's q/k/v channel group -> per-head
// LayerNorm in-register; packed bf16 (B,HE,HW,HD) outputs.
// ---------------------------------------------------------------------------
__global__ __launch_bounds__(256) void gemm_qkv_kernel(
    const __bf16* __restrict__ A, const __bf16* __restrict__ Bt,
    const float* __restrict__ bias,
    const float* __restrict__ qsc, const float* __restrict__ qbi,
    const float* __restrict__ ksc, const float* __restrict__ kbi,
    __bf16* __restrict__ qo, __bf16* __restrict__ ko, __bf16* __restrict__ vo) {
  const int K = CH;
  __shared__ __bf16 As[2][128 * GSTR];
  __shared__ __bf16 Bs[2][128 * GSTR];
  const int tid  = threadIdx.x;
  const int lane = tid & 31;
  const int wv   = tid >> 5;
  const int wm   = wv & 3;
  const int wn   = wv >> 2;
  const int m0   = blockIdx.y * 128;
  const int n0   = blockIdx.x * 128;

  auto issue_tile = [&](int k0, int buf) {
#pragma unroll
    for (int i = 0; i < 4; ++i) {
      const int ch = tid + i * 256;
      const int r = ch >> 3, cc = ch & 7;
      cp16(A  + (long)(m0 + r) * K + k0 + cc * 8, &As[buf][r * GSTR + cc * 8]);
      cp16(Bt + (long)(n0 + r) * K + k0 + cc * 8, &Bs[buf][r * GSTR + cc * 8]);
    }
  };

  const v8f vz = {0.f, 0.f, 0.f, 0.f, 0.f, 0.f, 0.f, 0.f};
  v8f acc[2][4];
#pragma unroll
  for (int s = 0; s < 2; ++s)
#pragma unroll
    for (int t = 0; t < 4; ++t) acc[s][t] = vz;

  const int KT = K / BKK;
  issue_tile(0, 0);
#pragma clang loop unroll(disable)
  for (int kt = 0; kt < KT; ++kt) {
    const int cur = kt & 1;
    if (kt + 1 < KT) {
      issue_tile((kt + 1) * BKK, cur ^ 1);
      ASYNC_WAIT(8);
    } else {
      ASYNC_WAIT(0);
    }
    __syncthreads();
#pragma unroll
    for (int ks = 0; ks < 2; ++ks) {
      const int ko = ks * 32;
      v16bf a0 = load_a_frag(&As[cur][(wm * 32) * GSTR + ko], GSTR, lane);
      v16bf a1 = load_a_frag(&As[cur][(wm * 32 + 16) * GSTR + ko], GSTR, lane);
#pragma unroll
      for (int t = 0; t < 4; ++t) {
        v16bf b = load_bt_frag(&Bs[cur][(wn * 64 + t * 16) * GSTR + ko], GSTR, lane);
        acc[0][t] = wmma_bf16(a0, b, acc[0][t]);
        acc[1][t] = wmma_bf16(a1, b, acc[1][t]);
      }
    }
    __syncthreads();
  }

  const int grp  = (n0 + wn * 64) >> 6;    // 64-channel group: head*3 + type
  const int head = grp / 3;
  const int type = grp % 3;
  __bf16* dst = (type == 0) ? qo : (type == 1) ? ko : vo;

#pragma unroll
  for (int s = 0; s < 2; ++s)
#pragma unroll
    for (int j = 0; j < 8; ++j) {
      const int tok = m0 + wm * 32 + s * 16 + j + ((lane & 16) ? 8 : 0);
      const int bb  = tok >> 12;
      const int hw  = tok & 4095;
      float v[4], s1 = 0.f, s2 = 0.f;
#pragma unroll
      for (int t = 0; t < 4; ++t) {
        const int n = n0 + wn * 64 + t * 16 + (lane & 15);
        v[t] = acc[s][t][j] + bias[n];
        s1 += v[t]; s2 += v[t] * v[t];
      }
      s1 = hsum16(s1); s2 = hsum16(s2);
      const float mean = s1 * (1.f / 64.f);
      const float rstd = rsqrtf(fmaxf(s2 * (1.f / 64.f) - mean * mean, 0.f) + 1e-6f);
      const long rowbase = ((long)(bb * NHE + head) * HWSP + hw) * HDIM;
#pragma unroll
      for (int t = 0; t < 4; ++t) {
        const int d = t * 16 + (lane & 15);
        float y;
        if (type == 0)      y = (v[t] - mean) * rstd * qsc[d] + qbi[d];
        else if (type == 1) y = (v[t] - mean) * rstd * ksc[d] + kbi[d];
        else                y = v[t];
        dst[rowbase + d] = f2bf(y);
      }
    }
}

// ---------------------------------------------------------------------------
// Fused axial attention: one workgroup (4 waves) per (b, he, line).
// AXIS 0: attend over W (line = h).  AXIS 1: attend over H (line = w).
// ACCUM 1: out += o (Y pass adds onto X pass).
// ---------------------------------------------------------------------------
template<int AXIS, int ACCUM>
__global__ __launch_bounds__(128) void attn_kernel(
    const __bf16* __restrict__ Q, const __bf16* __restrict__ K,
    const __bf16* __restrict__ V, const float* __restrict__ bias_tab,
    float* __restrict__ out) {
  __shared__ __bf16 Qs[64 * LDQ];
  __shared__ __bf16 Ks[64 * LDQ];
  __shared__ __bf16 Vt[64 * LDQ];            // transposed: Vt[d][key]
  __shared__ __bf16 Ps[4 * 16 * LDQ];        // per-wave probs strip

  const int line = blockIdx.x;
  const int he   = blockIdx.y;
  const int b    = blockIdx.z;
  const int tid  = threadIdx.x;
  const int lane = tid & 31;
  const int wv   = tid >> 5;

  const long headbase = (long)(b * NHE + he) * HWSP * HDIM;
  const long base     = headbase + ((AXIS == 0) ? (long)line * 64 * 64
                                                : (long)line * 64);
  const long rstride  = (AXIS == 0) ? 64 : (long)64 * 64;

  for (int i = tid; i < 256; i += 128) {     // Q,K: 64 rows x 4 x 16B (async)
    const int r = i >> 2, c = i & 3;
    cp16(Q + base + r * rstride + c * 8, Qs + r * LDQ + c * 8);
    cp16(K + base + r * rstride + c * 8, Ks + r * LDQ + c * 8);
  }
  for (int i = tid; i < 2048; i += 128) {    // V transposed into LDS
    const int r  = i >> 5;
    const int c2 = (i & 31) * 2;
    const unsigned pv = *(const unsigned*)(V + base + r * rstride + c2);
    Vt[c2 * LDQ + r]       = __builtin_bit_cast(__bf16, (unsigned short)(pv & 0xffffu));
    Vt[(c2 + 1) * LDQ + r] = __builtin_bit_cast(__bf16, (unsigned short)(pv >> 16));
  }
  ASYNC_WAIT(0);
  __syncthreads();

  const v8f vz = {0.f, 0.f, 0.f, 0.f, 0.f, 0.f, 0.f, 0.f};
  v8f sacc[4];
#pragma unroll
  for (int t = 0; t < 4; ++t) sacc[t] = vz;
  {
    v16bf a0 = load_a_frag(Qs + wv * 16 * LDQ, LDQ, lane);        // K dims 0..31
    v16bf a1 = load_a_frag(Qs + wv * 16 * LDQ + 32, LDQ, lane);   // K dims 32..63
#pragma unroll
    for (int t = 0; t < 4; ++t) {
      v16bf b0 = load_bt_frag(Ks + t * 16 * LDQ, LDQ, lane);      // B = K^T
      v16bf b1 = load_bt_frag(Ks + t * 16 * LDQ + 32, LDQ, lane);
      sacc[t] = wmma_bf16(a0, b0, sacc[t]);
      sacc[t] = wmma_bf16(a1, b1, sacc[t]);
    }
  }

  const float* brow = bias_tab + he * 64 * 64;
  float rinv[8];
#pragma unroll
  for (int j = 0; j < 8; ++j) {
    const int ml   = j + ((lane & 16) ? 8 : 0);
    const int mrow = wv * 16 + ml;
    float sv[4];
    float mx = -3.0e38f;
#pragma unroll
    for (int t = 0; t < 4; ++t) {
      const int col = t * 16 + (lane & 15);
      sv[t] = sacc[t][j] * 0.125f + brow[mrow * 64 + col];        // scale = hd^-0.5
      mx = fmaxf(mx, sv[t]);
    }
    mx = hmax16(mx);
    float sum = 0.f;
#pragma unroll
    for (int t = 0; t < 4; ++t) { sv[t] = __expf(sv[t] - mx); sum += sv[t]; }
    sum = hsum16(sum);
    rinv[j] = 1.0f / sum;
#pragma unroll
    for (int t = 0; t < 4; ++t)
      Ps[wv * 16 * LDQ + ml * LDQ + t * 16 + (lane & 15)] = f2bf(sv[t]);
  }
  // Same-wave LDS write->read is ordered (DScnt in-order); no barrier needed.

  v8f oacc[4];
#pragma unroll
  for (int t = 0; t < 4; ++t) oacc[t] = vz;
  {
    v16bf p0 = load_a_frag(Ps + wv * 16 * LDQ, LDQ, lane);
    v16bf p1 = load_a_frag(Ps + wv * 16 * LDQ + 32, LDQ, lane);
#pragma unroll
    for (int t = 0; t < 4; ++t) {
      v16bf b0 = load_bt_frag(Vt + t * 16 * LDQ, LDQ, lane);      // B = V
      v16bf b1 = load_bt_frag(Vt + t * 16 * LDQ + 32, LDQ, lane);
      oacc[t] = wmma_bf16(p0, b0, oacc[t]);
      oacc[t] = wmma_bf16(p1, b1, oacc[t]);
    }
  }

#pragma unroll
  for (int j = 0; j < 8; ++j) {
    const int ml   = j + ((lane & 16) ? 8 : 0);
    const int mrow = wv * 16 + ml;
#pragma unroll
    for (int t = 0; t < 4; ++t) {
      const int  d   = t * 16 + (lane & 15);
      const long idx = base + (long)mrow * rstride + d;
      const float val = oacc[t][j] * rinv[j];
      if (ACCUM) out[idx] += val; else out[idx] = val;
    }
  }
}

// ---------------------------------------------------------------------------
extern "C" void kernel_launch(void* const* d_in, const int* in_sizes, int n_in,
                              void* d_out, int out_size, void* d_ws, size_t ws_size,
                              hipStream_t stream) {
  const float* x         = (const float*)d_in[0];
  const float* norm1_w   = (const float*)d_in[2];
  const float* w_qkv     = (const float*)d_in[3];
  const float* b_qkv     = (const float*)d_in[4];
  const float* q_scale   = (const float*)d_in[5];
  const float* q_bias    = (const float*)d_in[6];
  const float* k_scale   = (const float*)d_in[7];
  const float* k_bias    = (const float*)d_in[8];
  const float* rel_table = (const float*)d_in[9];
  const float* norm2_w   = (const float*)d_in[10];
  const float* w_out     = (const float*)d_in[11];
  const float* b_out     = (const float*)d_in[12];
  const float* gamma_att = (const float*)d_in[13];
  const float* w1        = (const float*)d_in[14];
  const float* b1        = (const float*)d_in[15];
  const float* w2        = (const float*)d_in[16];
  const float* b2        = (const float*)d_in[17];
  const float* mlp_norm  = (const float*)d_in[18];
  const float* gamma_mlp = (const float*)d_in[19];
  float* out = (float*)d_out;

  size_t off = 0;
  auto alloc = [&](size_t bytes) -> void* {
    void* p = (char*)d_ws + off;
    off += (bytes + 255) & ~(size_t)255;
    return p;
  };
  __bf16* t_bf    = (__bf16*)alloc((size_t)TOKS * CH * 2);       // reused as a_bf
  __bf16* q_bf    = (__bf16*)alloc((size_t)TOKS * CH * 2);
  __bf16* k_bf    = (__bf16*)alloc((size_t)TOKS * CH * 2);
  __bf16* v_bf    = (__bf16*)alloc((size_t)TOKS * CH * 2);
  float*  attn    = (float*)alloc((size_t)TOKS * CH * 4);        // reused as m
  float*  x1_f    = (float*)alloc((size_t)TOKS * CH * 4);
  __bf16* x1_bf   = (__bf16*)alloc((size_t)TOKS * CH * 2);
  __bf16* h_bf    = (__bf16*)alloc((size_t)TOKS * 4 * CH * 2);
  __bf16* wqkvT   = (__bf16*)alloc((size_t)CH * 3 * CH * 2);     // (3C x C)
  __bf16* woutT   = (__bf16*)alloc((size_t)CH * CH * 2);         // (C x C)
  __bf16* w1T     = (__bf16*)alloc((size_t)CH * 4 * CH * 2);     // (4C x C)
  __bf16* w2T     = (__bf16*)alloc((size_t)4 * CH * CH * 2);     // (C x 4C)
  float*  scale1  = (float*)alloc((size_t)NB * CH * 4);
  float*  scale2  = (float*)alloc((size_t)NB * CH * 4);
  float*  scale3  = (float*)alloc((size_t)NB * CH * 4);
  float*  biastab = (float*)alloc((size_t)NHE * 64 * 64 * 4);
  float*  part1   = (float*)alloc((size_t)SLABS * NB * CH * 4);
  float*  part2   = (float*)alloc((size_t)SLABS * NB * CH * 4);
  __bf16* a_bf  = t_bf;     // t dead after QKV GEMM
  float*  m_f   = attn;     // attn dead after norm2 apply

  const long n_elem = (long)TOKS * CH;          // 25,165,824
  const int  g_ew   = (int)(n_elem / 256);      // 98,304 blocks
  const dim3 gstat(CH / 256, NB, SLABS);
  const dim3 gfin(CH / 256, NB);

  // 1) weights -> transposed bf16 (n-major rows of length K)
  transpose_to_bf16_kernel<<<(CH * 3 * CH + 255) / 256, 256, 0, stream>>>(w_qkv, wqkvT, CH, 3 * CH);
  transpose_to_bf16_kernel<<<(CH * CH + 255) / 256, 256, 0, stream>>>(w_out, woutT, CH, CH);
  transpose_to_bf16_kernel<<<(CH * 4 * CH + 255) / 256, 256, 0, stream>>>(w1, w1T, CH, 4 * CH);
  transpose_to_bf16_kernel<<<(4 * CH * CH + 255) / 256, 256, 0, stream>>>(w2, w2T, 4 * CH, CH);
  // 2) relative position bias table
  build_bias_kernel<<<(NHE * 64 * 64) / 256, 256, 0, stream>>>(rel_table, biastab);
  // 3) norm1 (instance rms) -> t_bf16
  inorm_partial_kernel<0><<<gstat, 256, 0, stream>>>(x, part1, part2, 1.0f);
  inorm_finish_kernel<<<gfin, 256, 0, stream>>>(part1, part2, norm1_w, scale1, 1.0f, 1e-8f);
  inorm_apply_kernel<0><<<g_ew, 256, 0, stream>>>(x, scale1, t_bf);
  // 4) QKV GEMM + per-head LN, packed (B,HE,HW,HD) bf16 q/k/v
  gemm_qkv_kernel<<<dim3(3 * CH / 128, TOKS / 128), 256, 0, stream>>>(
      t_bf, wqkvT, b_qkv, q_scale, q_bias, k_scale, k_bias, q_bf, k_bf, v_bf);
  // 5) axial attention: X pass writes, Y pass accumulates
  attn_kernel<0, 0><<<dim3(64, NHE, NB), 128, 0, stream>>>(q_bf, k_bf, v_bf, biastab, attn);
  attn_kernel<1, 1><<<dim3(64, NHE, NB), 128, 0, stream>>>(q_bf, k_bf, v_bf, biastab, attn);
  // 6) norm2 on a = 0.5*(xx+xy) (pre=0.5 folded into scale)
  inorm_partial_kernel<1><<<gstat, 256, 0, stream>>>(attn, part1, part2, 0.5f);
  inorm_finish_kernel<<<gfin, 256, 0, stream>>>(part1, part2, norm2_w, scale2, 0.5f, 1e-8f);
  inorm_apply_kernel<1><<<g_ew, 256, 0, stream>>>(attn, scale2, a_bf);
  // 7) out proj + layer scale + residual -> x1 (f32 + bf16)
  gemm_bf16_kernel<2><<<dim3(CH / 128, TOKS / 128), 256, 0, stream>>>(
      a_bf, woutT, b_out, gamma_att, x, x1_f, x1_bf, TOKS, CH, CH);
  // 8) MLP
  gemm_bf16_kernel<1><<<dim3(4 * CH / 128, TOKS / 128), 256, 0, stream>>>(
      x1_bf, w1T, b1, nullptr, nullptr, nullptr, h_bf, TOKS, 4 * CH, CH);
  gemm_bf16_kernel<0><<<dim3(CH / 128, TOKS / 128), 256, 0, stream>>>(
      h_bf, w2T, b2, nullptr, nullptr, m_f, nullptr, TOKS, CH, 4 * CH);
  // 9) mlp norm + layer scale + residual
  inorm_partial_kernel<0><<<gstat, 256, 0, stream>>>(m_f, part1, part2, 1.0f);
  inorm_finish_kernel<<<gfin, 256, 0, stream>>>(part1, part2, mlp_norm, scale3, 1.0f, 1e-8f);
  final_kernel<<<g_ew, 256, 0, stream>>>(m_f, scale3, gamma_mlp, x1_f, out);
}
